// Head_12738873000101
// MI455X (gfx1250) — matrix operand order repaired and checked by reading, not compiled, not run on hardware
//
#include <hip/hip_runtime.h>

// Problem constants (match reference)
#define B_    4
#define T_    4096
#define EMB_  1024
#define HEAD_ 64
#define NROW_ (B_ * T_)   // 16384 rows total

#define NSPLIT_ 4                 // split-j factor for attention parallelism
#define PSTRIDE_ (16 * HEAD_ + 32)  // per-(tile,split) partial: O[16][64] + M[16] + L[16]

typedef __attribute__((ext_vector_type(16))) _Float16 v16h;
typedef __attribute__((ext_vector_type(8)))  _Float16 v8h;
typedef __attribute__((ext_vector_type(8)))  float    v8f;
typedef __attribute__((ext_vector_type(4)))  float    v4f;

__device__ __forceinline__ v16h cat8(v8h lo, v8h hi) {
    return __builtin_shufflevector(lo, hi, 0, 1, 2, 3, 4, 5, 6, 7,
                                           8, 9, 10, 11, 12, 13, 14, 15);
}

// ---------------------------------------------------------------------------
// Kernel 0: one-time weight convert+transpose: Wt[wsel][d][k] f16.
// Makes proj B-fragments two contiguous 8-half runs (two b128 loads).
// ---------------------------------------------------------------------------
__global__ __launch_bounds__(256) void prep_w(
    const float* __restrict__ Wk,
    const float* __restrict__ Wq,
    const float* __restrict__ Wv,
    _Float16* __restrict__ Wt)
{
    int idx  = blockIdx.x * 256 + threadIdx.x;   // 0 .. 3*EMB*HEAD-1
    int wsel = idx / (EMB_ * HEAD_);             // uniform per block
    int rem  = idx % (EMB_ * HEAD_);
    int k    = rem >> 6;                         // row (coalesced read)
    int d    = rem & 63;
    const float* W = (wsel == 0) ? Wk : (wsel == 1) ? Wq : Wv;
    Wt[((size_t)wsel * HEAD_ + d) * EMB_ + k] = (_Float16)W[rem];
}

// ---------------------------------------------------------------------------
// Kernel 1: K/Q/V projections, LDS-free.
// Block = 4 waves; wave w owns output columns [16w,16w+16).
// A-fragment: x rows as f32 b128 loads + cvt; B-fragment: Wt f16 b128 loads.
// Outputs: Kp,Qp row-major [NROW,H] f16;  Vt transposed [B,H,T] f16.
// ---------------------------------------------------------------------------
__global__ __launch_bounds__(128) void proj_kernel(
    const float* __restrict__ x,
    const _Float16* __restrict__ Wt,   // [3][HEAD][EMB]
    _Float16* __restrict__ Kp,
    _Float16* __restrict__ Qp,
    _Float16* __restrict__ Vt)
{
    const int lane = threadIdx.x & 31;
    const int wav  = threadIdx.x >> 5;
    const int m    = lane & 15;
    const int h    = lane >> 4;
    const int row0 = blockIdx.x * 16;
    const int d    = wav * 16 + m;

    const float*    xrow = x  + (size_t)(row0 + m) * EMB_;
    const _Float16* wk   = Wt + ((size_t)0 * HEAD_ + d) * EMB_;
    const _Float16* wq   = Wt + ((size_t)1 * HEAD_ + d) * EMB_;
    const _Float16* wv   = Wt + ((size_t)2 * HEAD_ + d) * EMB_;

    v8f accK = {}, accQ = {}, accV = {};

    for (int k0 = 0; k0 < EMB_; k0 += 32) {
        if (k0 + 32 < EMB_)
            __builtin_prefetch(xrow + k0 + 32, 0, 3);   // global_prefetch_b8

        // ---- A fragment: lane half h holds K = h*8+0..7 and 16+h*8+0..7
        v4f xa0 = *(const v4f*)(xrow + k0 + h * 8);
        v4f xa1 = *(const v4f*)(xrow + k0 + h * 8 + 4);
        v4f xb0 = *(const v4f*)(xrow + k0 + 16 + h * 8);
        v4f xb1 = *(const v4f*)(xrow + k0 + 16 + h * 8 + 4);
        v16h a;
        #pragma unroll
        for (int i = 0; i < 4; ++i) {
            a[i]      = (_Float16)xa0[i];
            a[4 + i]  = (_Float16)xa1[i];
            a[8 + i]  = (_Float16)xb0[i];
            a[12 + i] = (_Float16)xb1[i];
        }

        // ---- B fragments: lane half h holds K = h*8+0..7 and 16+h*8+0..7
        v16h bk = cat8(*(const v8h*)(wk + k0 + h * 8), *(const v8h*)(wk + k0 + 16 + h * 8));
        v16h bq = cat8(*(const v8h*)(wq + k0 + h * 8), *(const v8h*)(wq + k0 + 16 + h * 8));
        v16h bv = cat8(*(const v8h*)(wv + k0 + h * 8), *(const v8h*)(wv + k0 + 16 + h * 8));

        accK = __builtin_amdgcn_wmma_f32_16x16x32_f16(false, a, false, bk, (short)0, accK, false, false);
        accQ = __builtin_amdgcn_wmma_f32_16x16x32_f16(false, a, false, bq, (short)0, accQ, false, false);
        accV = __builtin_amdgcn_wmma_f32_16x16x32_f16(false, a, false, bv, (short)0, accV, false, false);
    }

    // C layout: element r -> row r+8h, col = m (within this wave's d tile)
    #pragma unroll
    for (int r = 0; r < 8; ++r) {
        int grow = row0 + r + 8 * h;
        Kp[(size_t)grow * HEAD_ + d] = (_Float16)accK[r];
        Qp[(size_t)grow * HEAD_ + d] = (_Float16)accQ[r];
        int b = grow >> 12;          // /T_
        int t = grow & (T_ - 1);
        Vt[((size_t)b * HEAD_ + d) * T_ + t] = (_Float16)accV[r];
    }
}

// ---------------------------------------------------------------------------
// Kernel 2: causal flash attention, split-j partials.
// Reference does w = k @ q^T, so K rows act as queries.  One wave per
// (16-row tile, split) pair; split s handles j-blocks j0 = 32*s, step 128.
// Emits raw partials (O_raw, M, L) per split; merged by merge_kernel.
// ---------------------------------------------------------------------------
__global__ __launch_bounds__(128) void attn_partial(
    const _Float16* __restrict__ Kp,   // [NROW, H] : row role
    const _Float16* __restrict__ Qp,   // [NROW, H] : col role
    const _Float16* __restrict__ Vt,   // [B, H, T]
    float* __restrict__ part)          // [NROW/16][NSPLIT][PSTRIDE_]
{
    __shared__ _Float16 pls[4][16][32];   // per-wave P staging, 64B rows (aligned)

    const int lane = threadIdx.x & 31;
    const int wav  = threadIdx.x >> 5;
    const int m    = lane & 15;
    const int h    = lane >> 4;
    const int job  = blockIdx.x * 4 + wav;   // 0 .. NROW_/16 * NSPLIT_ - 1
    const int tile = job >> 2;               // / NSPLIT_
    const int spl  = job & (NSPLIT_ - 1);
    const int b    = tile >> 8;              // / (T_/16)
    const int i0   = (tile & 255) << 4;

    const _Float16* Krow = Kp + ((size_t)b * T_ + i0) * HEAD_;
    const _Float16* Qb   = Qp + (size_t)b * T_ * HEAD_;
    const _Float16* Vb   = Vt + (size_t)b * HEAD_ * T_;

    // A fragments of the K row tile (M=i, K=d), two 32-wide d chunks
    v16h aK[2];
    #pragma unroll
    for (int c = 0; c < 2; ++c) {
        const _Float16* base = Krow + (size_t)m * HEAD_ + c * 32;
        aK[c] = cat8(*(const v8h*)(base + h * 16),
                     *(const v8h*)(base + h * 16 + 8));
    }

    const float scale = 1.0f / ((float)EMB_ * (float)EMB_);
    float Mrow[8], Lrow[8];
    v8f O[4] = {};
    #pragma unroll
    for (int r = 0; r < 8; ++r) { Mrow[r] = -__builtin_inff(); Lrow[r] = 0.0f; }

    const int imax = i0 + 15;
    const int jstep = 32 * NSPLIT_;
    for (int j0 = 32 * spl; j0 <= imax; j0 += jstep) {
        if (j0 + jstep <= imax) {   // warm this split's next j-block (Q + V rows)
            __builtin_prefetch(Qb + (size_t)(j0 + jstep + lane) * HEAD_, 0, 3);
            __builtin_prefetch(Vb + (size_t)lane * T_ + j0 + jstep, 0, 3);
            __builtin_prefetch(Vb + (size_t)(32 + lane) * T_ + j0 + jstep, 0, 3);
        }
        // ---- S = Krow @ Q^T for this 16x32 j-block (2 col tiles x 2 k chunks)
        v8f S[2];
        #pragma unroll
        for (int ct = 0; ct < 2; ++ct) {
            v8f s = {};
            #pragma unroll
            for (int c = 0; c < 2; ++c) {
                const _Float16* base = Qb + (size_t)(j0 + ct * 16 + m) * HEAD_ + c * 32 + h * 16;
                v16h bQ = cat8(*(const v8h*)(base), *(const v8h*)(base + 8));
                s = __builtin_amdgcn_wmma_f32_16x16x32_f16(false, aK[c], false, bQ, (short)0, s, false, false);
            }
            S[ct] = s;
        }
        // ---- causal mask + 1/C^2 scale (C layout: row=i0+r+8h, col=j0+ct*16+m)
        #pragma unroll
        for (int ct = 0; ct < 2; ++ct) {
            int j = j0 + ct * 16 + m;
            #pragma unroll
            for (int r = 0; r < 8; ++r) {
                int i = i0 + r + 8 * h;
                float v = S[ct][r] * scale;
                S[ct][r] = (j <= i) ? v : -__builtin_inff();
            }
        }
        // ---- online softmax (row reductions across 16-lane halves)
        float alpha[8];
        #pragma unroll
        for (int r = 0; r < 8; ++r) {
            float rm = fmaxf(S[0][r], S[1][r]);
            #pragma unroll
            for (int msk = 8; msk >= 1; msk >>= 1)
                rm = fmaxf(rm, __shfl_xor(rm, msk, 32));
            float mn = fmaxf(Mrow[r], rm);
            alpha[r] = __expf(Mrow[r] - mn);
            Mrow[r]  = mn;
            float p0 = __expf(S[0][r] - mn);
            float p1 = __expf(S[1][r] - mn);
            S[0][r] = p0; S[1][r] = p1;
            float ps = p0 + p1;
            #pragma unroll
            for (int msk = 8; msk >= 1; msk >>= 1)
                ps += __shfl_xor(ps, msk, 32);
            Lrow[r] = Lrow[r] * alpha[r] + ps;
        }
        #pragma unroll
        for (int dt = 0; dt < 4; ++dt)
            #pragma unroll
            for (int r = 0; r < 8; ++r)
                O[dt][r] *= alpha[r];
        // ---- P: C layout -> A layout via per-wave LDS round trip
        #pragma unroll
        for (int ct = 0; ct < 2; ++ct)
            #pragma unroll
            for (int r = 0; r < 8; ++r)
                pls[wav][r + 8 * h][ct * 16 + m] = (_Float16)S[ct][r];
        v16h aP = cat8(*(const v8h*)&pls[wav][m][h * 16],
                       *(const v8h*)&pls[wav][m][h * 16 + 8]);
        // ---- O += P @ V   (V transposed so fragment runs are contiguous in T)
        #pragma unroll
        for (int dt = 0; dt < 4; ++dt) {
            const _Float16* base = Vb + (size_t)(dt * 16 + m) * T_ + j0 + h * 16;
            v16h bV = cat8(*(const v8h*)(base), *(const v8h*)(base + 8));
            O[dt] = __builtin_amdgcn_wmma_f32_16x16x32_f16(false, aP, false, bV, (short)0, O[dt], false, false);
        }
    }

    // ---- emit raw partials (not normalized; merge kernel rescales)
    float* P = part + (size_t)job * PSTRIDE_;
    #pragma unroll
    for (int r = 0; r < 8; ++r) {
        int row = r + 8 * h;
        #pragma unroll
        for (int dt = 0; dt < 4; ++dt)
            P[row * HEAD_ + dt * 16 + m] = O[dt][r];
        if (m == 0) {                         // lanes 0 and 16 own the stats
            P[16 * HEAD_ + row]      = Mrow[r];
            P[16 * HEAD_ + 16 + row] = Lrow[r];
        }
    }
}

// ---------------------------------------------------------------------------
// Kernel 3: merge split-j partials with log-sum-exp rescaling.
// One thread per output element.
// ---------------------------------------------------------------------------
__global__ __launch_bounds__(256) void merge_kernel(
    const float* __restrict__ part,
    float* __restrict__ out)
{
    int idx  = blockIdx.x * 256 + threadIdx.x;   // 0 .. NROW_*HEAD_-1
    int row  = idx >> 6;
    int d    = idx & 63;
    int tile = row >> 4;
    int r    = row & 15;

    float Ms[NSPLIT_], Ls[NSPLIT_], Os[NSPLIT_];
    float mstar = -__builtin_inff();
    #pragma unroll
    for (int s = 0; s < NSPLIT_; ++s) {
        const float* P = part + ((size_t)tile * NSPLIT_ + s) * PSTRIDE_;
        Ms[s] = P[16 * HEAD_ + r];
        Ls[s] = P[16 * HEAD_ + 16 + r];
        Os[s] = P[r * HEAD_ + d];
        mstar = fmaxf(mstar, Ms[s]);
    }
    float num = 0.0f, den = 0.0f;
    #pragma unroll
    for (int s = 0; s < NSPLIT_; ++s) {
        float w = __expf(Ms[s] - mstar);   // 0 for empty splits (M=-inf)
        num += w * Os[s];
        den += w * Ls[s];
    }
    out[idx] = num / den;
}

extern "C" void kernel_launch(void* const* d_in, const int* in_sizes, int n_in,
                              void* d_out, int out_size, void* d_ws, size_t ws_size,
                              hipStream_t stream) {
    const float* x  = (const float*)d_in[0];
    const float* Wk = (const float*)d_in[1];
    const float* Wq = (const float*)d_in[2];
    const float* Wv = (const float*)d_in[3];
    float* out = (float*)d_out;

    _Float16* Kp = (_Float16*)d_ws;                       // 2 MB
    _Float16* Qp = Kp + (size_t)NROW_ * HEAD_;            // 2 MB
    _Float16* Vt = Qp + (size_t)NROW_ * HEAD_;            // 2 MB
    _Float16* Wt = Vt + (size_t)NROW_ * HEAD_;            // 384 KB
    float*  part = (float*)(Wt + (size_t)3 * HEAD_ * EMB_);  // ~17.3 MB

    prep_w<<<(3 * EMB_ * HEAD_) / 256, 256, 0, stream>>>(Wk, Wq, Wv, Wt);
    proj_kernel<<<NROW_ / 16, 128, 0, stream>>>(x, Wt, Kp, Qp, Vt);
    attn_partial<<<(NROW_ / 16) * NSPLIT_ / 4, 128, 0, stream>>>(Kp, Qp, Vt, part);
    merge_kernel<<<(NROW_ * HEAD_) / 256, 256, 0, stream>>>(part, out);
}
